// MeshMultiHeadAttention_6408091206165
// MI455X (gfx1250) — compile-verified, matrix operand order checked by reading, not run on hardware
//
#include <hip/hip_runtime.h>
#include <math.h>

// ---------------------------------------------------------------------------
// MeshMultiHeadAttention for MI455X (gfx1250, wave32, WMMA).
// - all matmuls through v_wmma_f32_16x16x32_bf16
// - 16B-aligned LDS tiles, uint4/float4 fragment loads (ds_load_b128 paths)
// - double-buffered staging; K tiles via global_load_async_to_lds_b64
// - softmax row reductions via DPP16 row_xmask (VALU only)
// - lift GEMMs consume producer-transposed bf16 copies (contiguous B frags)
// ---------------------------------------------------------------------------

typedef __attribute__((ext_vector_type(16))) __bf16 v16bf;
typedef __attribute__((ext_vector_type(8)))  float  v8f;

#define DMODEL 256   // model width (= H * DK = 8 * 32)

static __device__ inline v8f wmma_bf16(v16bf a, v16bf b, v8f c) {
  return __builtin_amdgcn_wmma_f32_16x16x32_bf16(
      false, a, false, b, (short)0, c, false, false);
}

// 16-lane xor butterfly step via DPP16 ROW_XMASK (VALU only)
template <int M>
static __device__ inline float bfly(float x) {
  const int v = __builtin_amdgcn_update_dpp(
      0, __builtin_bit_cast(int, x), 0x160 | M, 0xF, 0xF, false);
  return __builtin_bit_cast(float, v);
}
static __device__ inline float grp16_max(float t) {
  t = fmaxf(t, bfly<1>(t)); t = fmaxf(t, bfly<2>(t));
  t = fmaxf(t, bfly<4>(t)); t = fmaxf(t, bfly<8>(t));
  return t;
}
static __device__ inline float grp16_sum(float t) {
  t += bfly<1>(t); t += bfly<2>(t); t += bfly<4>(t); t += bfly<8>(t);
  return t;
}

// --- Fragment loaders --------------------------------------------------------
// 16x32 bf16 operand, wave32 layout: lane = 16*h + r
//   elems 0..7 -> k = k0 + 8h + i ; elems 8..15 -> k = k0 + 16 + 8h + i
// All sources are 16B aligned at (row*ld + k0 + 8h) -> two 16B vector loads.

struct frag_bits { uint4 lo, hi; };

static __device__ inline v16bf frag_row_bf16(const __bf16* __restrict__ p, int ld,
                                             int row0, int k0, int lane) {
  const int r = lane & 15, h = lane >> 4;
  const __bf16* q = p + (size_t)(row0 + r) * ld + k0 + h * 8;
  frag_bits s;
  s.lo = *(const uint4*)q;
  s.hi = *(const uint4*)(q + 16);
  return __builtin_bit_cast(v16bf, s);
}

static __device__ inline v16bf frag_row_f32(const float* __restrict__ p, int ld,
                                            int row0, int k0, int lane) {
  const int r = lane & 15, h = lane >> 4;
  const float* q = p + (size_t)(row0 + r) * ld + k0 + h * 8;
  const float4 x0 = *(const float4*)(q);
  const float4 x1 = *(const float4*)(q + 4);
  const float4 y0 = *(const float4*)(q + 16);
  const float4 y1 = *(const float4*)(q + 20);
  v16bf f;
  f[0] = (__bf16)x0.x; f[1]  = (__bf16)x0.y; f[2]  = (__bf16)x0.z; f[3]  = (__bf16)x0.w;
  f[4] = (__bf16)x1.x; f[5]  = (__bf16)x1.y; f[6]  = (__bf16)x1.z; f[7]  = (__bf16)x1.w;
  f[8] = (__bf16)y0.x; f[9]  = (__bf16)y0.y; f[10] = (__bf16)y0.z; f[11] = (__bf16)y0.w;
  f[12] = (__bf16)y1.x; f[13] = (__bf16)y1.y; f[14] = (__bf16)y1.z; f[15] = (__bf16)y1.w;
  return f;
}

// ---------------------------------------------------------------------------
// WMMA GEMM: C[M,256] = A[M,K] @ B^T, B stored [256, K]-style row-major
// (weights, or producer-transposed bf16 copies for the d0/d1 lifts).
// Block = 8 waves = one 16-row x 256-col stripe; A tile staged once in LDS
// (double-buffered) and shared by all 8 waves.
//   FUSE_LN : per-head LayerNorm epilogue (gamma/beta flat[256]) -> bf16
//   OUT_F32 : float output + bias (final projections)
//   A_SUM   : A := A + A2 elementwise (xe_f + xe_v)
//   STORE_T : also write transposed bf16 copy T[col][row] (lift inputs)
// Grid: x = M/16, y = batch.
// ---------------------------------------------------------------------------
template <bool A_F32, bool B_F32, bool FUSE_LN, bool OUT_F32, bool FUSE_BIAS,
          bool A_SUM, bool STORE_T>
__global__ __launch_bounds__(256) void gemm_wmma(
    const void* __restrict__ Ap, const void* __restrict__ A2p,
    const void* __restrict__ Bp,
    const float* __restrict__ gamma, const float* __restrict__ beta,
    const float* __restrict__ bias,
    void* __restrict__ Cp, __bf16* __restrict__ Tp,
    int Kdim, int ldb, int ldt,
    long long asb, long long bsb, long long csb) {
  const int tid = threadIdx.x, lane = tid & 31, wave = tid >> 5;
  const int row0 = blockIdx.x * 16;
  const int col0 = wave * 32;
  const long long b = blockIdx.y;

  const float*  Af  = (const float*)Ap;
  const __bf16* Ab  = (const __bf16*)Ap;
  const __bf16* A2b = (const __bf16*)A2p;
  const float*  Bf  = (const float*)Bp;
  const __bf16* Bb  = (const __bf16*)Bp;

  __shared__ alignas(16) __bf16 lds_a[2][16][40];        // double-buffered A tile
  __shared__ float lnb[FUSE_LN ? 8 : 1][16][33];         // LN staging

  const int ar = tid >> 4;        // stage: row 0..15
  const int ac = (tid & 15) * 2;  // stage: 2 consecutive k

  auto stageA = [&](int k0, int buf) {
    unsigned pk;
    if constexpr (A_F32) {
      const float* A = Af + b * asb + (size_t)(row0 + ar) * Kdim + k0 + ac;
      __builtin_prefetch(A + 64, 0, 1);
      const float2 av = *(const float2*)A;
      union { __bf16 h[2]; unsigned u; } u;
      u.h[0] = (__bf16)av.x; u.h[1] = (__bf16)av.y;
      pk = u.u;
    } else {
      const __bf16* A = Ab + b * asb + (size_t)(row0 + ar) * Kdim + k0 + ac;
      if constexpr (A_SUM) {
        const __bf16* A2 = A2b + b * asb + (size_t)(row0 + ar) * Kdim + k0 + ac;
        union { __bf16 h[2]; unsigned u; } u;
        u.h[0] = (__bf16)((float)A[0] + (float)A2[0]);
        u.h[1] = (__bf16)((float)A[1] + (float)A2[1]);
        pk = u.u;
      } else {
        pk = *(const unsigned*)A;
      }
    }
    *(unsigned*)&lds_a[buf][ar][ac] = pk;
  };

  v8f acc0 = {0, 0, 0, 0, 0, 0, 0, 0};
  v8f acc1 = {0, 0, 0, 0, 0, 0, 0, 0};

  const int nk = Kdim >> 5;
  stageA(0, 0);
  int cur = 0;
  for (int kt = 0; kt < nk; ++kt) {
    __syncthreads();  // tile kt staged; previous tile's reads complete
    if (kt + 1 < nk) stageA((kt + 1) << 5, cur ^ 1);
    const int k0 = kt << 5;
    const v16bf a = frag_row_bf16(&lds_a[cur][0][0], 40, 0, 0, lane);
    v16bf b0, b1;
    if constexpr (B_F32) {
      b0 = frag_row_f32(Bf + b * bsb, ldb, col0, k0, lane);
      b1 = frag_row_f32(Bf + b * bsb, ldb, col0 + 16, k0, lane);
    } else {
      b0 = frag_row_bf16(Bb + b * bsb, ldb, col0, k0, lane);
      b1 = frag_row_bf16(Bb + b * bsb, ldb, col0 + 16, k0, lane);
    }
    acc0 = wmma_bf16(a, b0, acc0);
    acc1 = wmma_bf16(a, b1, acc1);
    cur ^= 1;
  }

  // C layout: lane = 16*h + r holds col (col0 + r [+16]), rows v + 8h.
  const int r = lane & 15, hh = lane >> 4;
  if constexpr (FUSE_LN) {
    __syncthreads();
#pragma unroll
    for (int v = 0; v < 8; ++v) {
      lnb[wave][v + 8 * hh][r]      = acc0[v];
      lnb[wave][v + 8 * hh][16 + r] = acc1[v];
    }
    __syncthreads();
    if (lane < 16) {
      float x[32];
      float s = 0.f;
#pragma unroll
      for (int j = 0; j < 32; ++j) { x[j] = lnb[wave][lane][j]; s += x[j]; }
      const float mu = s * (1.f / 32.f);
      float s2 = 0.f;
#pragma unroll
      for (int j = 0; j < 32; ++j) { const float d = x[j] - mu; s2 += d * d; }
      const float rs = rsqrtf(s2 * (1.f / 32.f) + 1e-5f);
      __bf16* Co = (__bf16*)Cp + b * csb + (size_t)(row0 + lane) * DMODEL + col0;
#pragma unroll
      for (int j = 0; j < 32; ++j)
        Co[j] = (__bf16)((x[j] - mu) * rs * gamma[col0 + j] + beta[col0 + j]);
    }
  } else if constexpr (OUT_F32) {
    float* Co = (float*)Cp + b * csb;
    float bb0 = 0.f, bb1 = 0.f;
    if constexpr (FUSE_BIAS) { bb0 = bias[col0 + r]; bb1 = bias[col0 + 16 + r]; }
#pragma unroll
    for (int v = 0; v < 8; ++v) {
      const size_t row = row0 + v + 8 * hh;
      Co[row * DMODEL + col0 + r]      = acc0[v] + bb0;
      Co[row * DMODEL + col0 + 16 + r] = acc1[v] + bb1;
    }
  } else {
    __bf16* Co = (__bf16*)Cp + b * csb;
#pragma unroll
    for (int v = 0; v < 8; ++v) {
      const size_t row = row0 + v + 8 * hh;
      const __bf16 e0 = (__bf16)acc0[v];
      const __bf16 e1 = (__bf16)acc1[v];
      Co[row * DMODEL + col0 + r]      = e0;
      Co[row * DMODEL + col0 + 16 + r] = e1;
      if constexpr (STORE_T) {
        Tp[(size_t)(col0 + r) * ldt + row]      = e0;
        Tp[(size_t)(col0 + 16 + r) * ldt + row] = e1;
      }
    }
  }
}

// ---------------------------------------------------------------------------
// Flash attention: O = softmax(Q K^T / sqrt(32)) @ V, head dim 32.
// 8 waves/block share one (batch, head): K/V tiles double-buffered in LDS,
// K via GLOBAL_LOAD_ASYNC_TO_LDS_B64 (ASYNCcnt), V staged transposed so
// all fragment reads are contiguous ds_load_b128.
// Optional OT: transposed bf16 copy of O (lift input for att2).
// Grid: x = nq/16/8, y = head (8), z = batch (2). Block = 256 (8 waves).
// ---------------------------------------------------------------------------
__global__ __launch_bounds__(256) void flash_attn(
    const __bf16* __restrict__ Qg, const __bf16* __restrict__ Kg,
    const __bf16* __restrict__ Vg, __bf16* __restrict__ Og,
    __bf16* __restrict__ OT, int ldt, int nq, int nkv) {
  __builtin_amdgcn_s_wait_tensorcnt(0);
  const int tid = threadIdx.x, lane = tid & 31, wave = tid >> 5;
  const int head = blockIdx.y;
  const size_t bz = blockIdx.z;
  const int q0 = (blockIdx.x * 8 + wave) * 16;

  const __bf16* Q = Qg + bz * nq * DMODEL + head * 32;
  const __bf16* K = Kg + bz * nkv * DMODEL + head * 32;
  const __bf16* V = Vg + bz * nkv * DMODEL + head * 32;
  __bf16*       O = Og + bz * nq * DMODEL + head * 32;

  __shared__ alignas(16) __bf16 lds_k[2][32][40];   // K tile [kv][d]
  __shared__ alignas(16) __bf16 lds_vt[2][32][40];  // V tile transposed [d][kv]
  __shared__ alignas(16) __bf16 pbuf[8][16][40];    // wave-private P staging

  const v16bf aq = frag_row_bf16(Q, DMODEL, q0, 0, lane);

  v8f o0 = {0, 0, 0, 0, 0, 0, 0, 0};
  v8f o1 = {0, 0, 0, 0, 0, 0, 0, 0};
  float mrow[8], lrow[8];
#pragma unroll
  for (int v = 0; v < 8; ++v) { mrow[v] = -1e30f; lrow[v] = 0.f; }

  const float scale = 0.17677669529663687f;  // 1/sqrt(32)
  const int r = lane & 15, hh = lane >> 4;
  const int sr = tid >> 3;  // staging: kv row 0..31
  const int sc = tid & 7;   // staging: 4-elem chunk 0..7
  // LDS byte offset = low 32 bits of the generic address (ISA 10.2 truncation)
  const unsigned kbase = (unsigned)(uintptr_t)&lds_k[0][0][0];
  const int KTILE_B = 32 * 40 * 2;

  auto stage = [&](int kvs, int buf) {
    // async-stage K tile: each thread moves 8B global -> LDS (ASYNCcnt)
    const unsigned ldsa = kbase + (unsigned)(buf * KTILE_B + sr * 80 + sc * 8);
    const unsigned long long ga =
        (unsigned long long)(uintptr_t)(K + (size_t)(kvs + sr) * DMODEL + sc * 4);
    asm volatile("global_load_async_to_lds_b64 %0, %1, off"
                 :: "v"(ldsa), "v"(ga) : "memory");
    // stage V tile transposed: [kv][d] -> [d][kv]
    const uint2 raw = *(const uint2*)(V + (size_t)(kvs + sr) * DMODEL + sc * 4);
    const __bf16* e = (const __bf16*)&raw;
#pragma unroll
    for (int j = 0; j < 4; ++j) lds_vt[buf][sc * 4 + j][sr] = e[j];
  };

  stage(0, 0);
  int cur = 0;
  for (int kv0 = 0; kv0 < nkv; kv0 += 32) {
    asm volatile("s_wait_asynccnt 0x0" ::: "memory");
    __syncthreads();  // tile kv0 staged everywhere; old buffer free
    if (kv0 + 32 < nkv) stage(kv0 + 32, cur ^ 1);  // overlap with compute

    // ---- S = Q K^T ----
    const v16bf bk0 = frag_row_bf16(&lds_k[cur][0][0], 40, 0, 0, lane);
    const v16bf bk1 = frag_row_bf16(&lds_k[cur][0][0], 40, 16, 0, lane);
    const v8f z = {0, 0, 0, 0, 0, 0, 0, 0};
    v8f s0 = wmma_bf16(aq, bk0, z);
    v8f s1 = wmma_bf16(aq, bk1, z);

    // ---- online softmax (row stats via DPP16 xor butterflies) ----
    float p0[8], p1[8];
#pragma unroll
    for (int v = 0; v < 8; ++v) {
      const float a = s0[v] * scale;
      const float c = s1[v] * scale;
      const float t = grp16_max(fmaxf(a, c));
      const float mn = fmaxf(mrow[v], t);
      const float alpha = __expf(mrow[v] - mn);
      mrow[v] = mn;
      p0[v] = __expf(a - mn);
      p1[v] = __expf(c - mn);
      const float rsum = grp16_sum(p0[v] + p1[v]);
      lrow[v] = lrow[v] * alpha + rsum;
      o0[v] *= alpha;
      o1[v] *= alpha;
    }

    // ---- restage P into A-operand layout (wave-private: LDS in-order) ----
#pragma unroll
    for (int v = 0; v < 8; ++v) {
      pbuf[wave][v + 8 * hh][r]      = (__bf16)p0[v];
      pbuf[wave][v + 8 * hh][16 + r] = (__bf16)p1[v];
    }
    const v16bf ap = frag_row_bf16(&pbuf[wave][0][0], 40, 0, 0, lane);

    // ---- O += P V ----
    const v16bf bv0 = frag_row_bf16(&lds_vt[cur][0][0], 40, 0, 0, lane);
    const v16bf bv1 = frag_row_bf16(&lds_vt[cur][0][0], 40, 16, 0, lane);
    o0 = wmma_bf16(ap, bv0, o0);
    o1 = wmma_bf16(ap, bv1, o1);
    cur ^= 1;
  }

#pragma unroll
  for (int v = 0; v < 8; ++v) {
    const float inv = 1.0f / lrow[v];
    const size_t row = q0 + v + 8 * hh;
    const __bf16 e0 = (__bf16)(o0[v] * inv);
    const __bf16 e1 = (__bf16)(o1[v] * inv);
    O[row * DMODEL + r]      = e0;
    O[row * DMODEL + 16 + r] = e1;
    if (OT) {  // transposed copy for the following lift GEMM
      __bf16* T = OT + bz * nq;
      T[(size_t)(head * 32 + r) * ldt + row]      = e0;
      T[(size_t)(head * 32 + 16 + r) * ldt + row] = e1;
    }
  }
}

// ---------------------------------------------------------------------------
// Host launcher
// ---------------------------------------------------------------------------
extern "C" void kernel_launch(void* const* d_in, const int* in_sizes, int n_in,
                              void* d_out, int out_size, void* d_ws, size_t ws_size,
                              hipStream_t stream) {
  (void)in_sizes; (void)n_in; (void)out_size; (void)ws_size;

  const float* x_v   = (const float*)d_in[0];
  const float* x_e   = (const float*)d_in[1];
  const float* x_f   = (const float*)d_in[2];
  const float* d0    = (const float*)d_in[3];
  const float* d1    = (const float*)d_in[4];
  const float* vW_Q  = (const float*)d_in[5];
  const float* vW_K  = (const float*)d_in[6];
  const float* vW_V  = (const float*)d_in[7];
  const float* eW_Q  = (const float*)d_in[8];
  const float* eW_K  = (const float*)d_in[9];
  const float* eW_V  = (const float*)d_in[10];
  const float* fW_K  = (const float*)d_in[11];
  const float* vW_o  = (const float*)d_in[12];
  const float* vb_o  = (const float*)d_in[13];
  const float* eW_o  = (const float*)d_in[14];
  const float* eb_o  = (const float*)d_in[15];
  const float* g_vQ  = (const float*)d_in[16]; const float* b_vQ = (const float*)d_in[17];
  const float* g_vK  = (const float*)d_in[18]; const float* b_vK = (const float*)d_in[19];
  const float* g_eQ  = (const float*)d_in[20]; const float* b_eQ = (const float*)d_in[21];
  const float* g_eK  = (const float*)d_in[22]; const float* b_eK = (const float*)d_in[23];
  const float* g_fK  = (const float*)d_in[24]; const float* b_fK = (const float*)d_in[25];

  const int BSz = 2, NV = 1024, ME = 3072, KF = 2048;
  const size_t SZV = (size_t)BSz * NV * DMODEL;  // 524288
  const size_t SZE = (size_t)BSz * ME * DMODEL;  // 1572864
  const size_t SZF = (size_t)BSz * KF * DMODEL;  // 1048576

  __bf16* ws = (__bf16*)d_ws;
  size_t off = 0;
  __bf16* vQ     = ws + off; off += SZV;
  __bf16* vK     = ws + off; off += SZV;
  __bf16* vV     = ws + off; off += SZV;
  __bf16* vVT    = ws + off; off += SZV;  // vV transposed [256][2048]
  __bf16* eQ     = ws + off; off += SZE;
  __bf16* eK     = ws + off; off += SZE;
  __bf16* eV     = ws + off; off += SZE;
  __bf16* eVT    = ws + off; off += SZE;  // eV transposed [256][6144]
  __bf16* fK     = ws + off; off += SZF;
  __bf16* liftVV = ws + off; off += SZE;  // d0 @ vV     [b, 3072, 256]
  __bf16* liftEV = ws + off; off += SZF;  // d1 @ eV     [b, 2048, 256]
  __bf16* xv     = ws + off; off += SZV;  // attn out v  [b, 1024, 256]
  __bf16* xef    = ws + off; off += SZE;  // attn out ef [b, 3072, 256]
  __bf16* att2   = ws + off; off += SZV;  // attn out v2 [b, 1024, 256]
  __bf16* att2T  = ws + off; off += SZV;  // att2 transposed [256][2048]
  __bf16* lift2  = ws + off; off += SZE;  // d0 @ att2   [b, 3072, 256]

  // ---- Stage A: projections (batch flattened into rows), K = 256 ----------
#define PROJ_LN(X, W, G, Bv, C, M)                                              \
  gemm_wmma<true, true, true, false, false, false, false>                       \
      <<<dim3((M) / 16, 1), 256, 0, stream>>>(X, nullptr, W, G, Bv, nullptr,    \
                                              C, nullptr, DMODEL, DMODEL, 0, 0, 0, 0)
#define PROJ_T(X, W, C, T, M)                                                   \
  gemm_wmma<true, true, false, false, false, false, true>                       \
      <<<dim3((M) / 16, 1), 256, 0, stream>>>(X, nullptr, W, nullptr, nullptr,  \
                                              nullptr, C, T, DMODEL, DMODEL, (M), 0, 0, 0)

  PROJ_LN(x_v, vW_Q, g_vQ, b_vQ, vQ, BSz * NV);
  PROJ_LN(x_v, vW_K, g_vK, b_vK, vK, BSz * NV);
  PROJ_T (x_v, vW_V, vV, vVT, BSz * NV);
  PROJ_LN(x_e, eW_Q, g_eQ, b_eQ, eQ, BSz * ME);
  PROJ_LN(x_e, eW_K, g_eK, b_eK, eK, BSz * ME);
  PROJ_T (x_e, eW_V, eV, eVT, BSz * ME);
  PROJ_LN(x_f, fW_K, g_fK, b_fK, fK, BSz * KF);

  // ---- Stage B: boundary lifts (A f32 +-1 matrix, B = transposed bf16) -----
  // liftVV[b] = d0[b] (3072x1024) @ vV[b]; B frag rows from vVT (ld 2048)
  gemm_wmma<true, false, false, false, false, false, false>
      <<<dim3(ME / 16, BSz), 256, 0, stream>>>(
          d0, nullptr, vVT, nullptr, nullptr, nullptr, liftVV, nullptr,
          NV, BSz * NV, 0, (long long)ME * NV, NV, (long long)ME * DMODEL);
  // liftEV[b] = d1[b] (2048x3072) @ eV[b]; B frag rows from eVT (ld 6144)
  gemm_wmma<true, false, false, false, false, false, false>
      <<<dim3(KF / 16, BSz), 256, 0, stream>>>(
          d1, nullptr, eVT, nullptr, nullptr, nullptr, liftEV, nullptr,
          ME, BSz * ME, 0, (long long)KF * ME, ME, (long long)KF * DMODEL);

  // ---- Stage C: flash attentions ------------------------------------------
  flash_attn<<<dim3(NV / 16 / 8, 8, BSz), 256, 0, stream>>>(
      vQ, eK, liftVV, xv, nullptr, 0, NV, ME);
  flash_attn<<<dim3(ME / 16 / 8, 8, BSz), 256, 0, stream>>>(
      eQ, fK, liftEV, xef, nullptr, 0, ME, KF);
  flash_attn<<<dim3(NV / 16 / 8, 8, BSz), 256, 0, stream>>>(
      vK, eK, eV, att2, att2T, BSz * NV, NV, ME);

  // lift2[b] = d0[b] @ att2[b]; B frag rows from att2T (ld 2048)
  gemm_wmma<true, false, false, false, false, false, false>
      <<<dim3(ME / 16, BSz), 256, 0, stream>>>(
          d0, nullptr, att2T, nullptr, nullptr, nullptr, lift2, nullptr,
          NV, BSz * NV, 0, (long long)ME * NV, NV, (long long)ME * DMODEL);

  // ---- Stage D: output projections (f32 out + bias) ------------------------
  float* out_v = (float*)d_out;
  float* out_e = (float*)d_out + SZV;
  gemm_wmma<false, true, false, true, true, false, false>
      <<<dim3(BSz * NV / 16, 1), 256, 0, stream>>>(
          xv, nullptr, vW_o, nullptr, nullptr, vb_o, out_v, nullptr,
          DMODEL, DMODEL, 0, 0, 0, 0);
  gemm_wmma<false, true, false, true, true, true, false>
      <<<dim3(BSz * ME / 16, 1), 256, 0, stream>>>(
          xef, lift2, eW_o, nullptr, nullptr, eb_o, out_e, nullptr,
          DMODEL, DMODEL, 0, 0, 0, 0);

#undef PROJ_LN
#undef PROJ_T
}
